// CausalSelfAttention_80384607912574
// MI455X (gfx1250) — matrix operand order repaired and checked by reading, not compiled, not run on hardware
//
#include <hip/hip_runtime.h>
#include <hip/hip_bf16.h>
#include <math.h>

// ---------------------------------------------------------------------------
// CDNA5 (gfx1250) causal self-attention, bf16 WMMA with fp32 accumulate.
// Pipeline: [qkv gemm] -> [flash attention] -> [proj gemm]
// Workspace layout (bf16): Q[BH,S,D] (pre-scaled by 0.125*log2e), K[BH,S,D],
//                          V^T[BH,D,S], O[B*S, C]. Total ~24 MB.
// ---------------------------------------------------------------------------

typedef __attribute__((ext_vector_type(16))) __bf16 v16bf;
typedef __attribute__((ext_vector_type(8)))  float  v8f;

__device__ __forceinline__ v8f wmma_bf16(v16bf a, v16bf b, v8f c) {
    // v_wmma_f32_16x16x32_bf16: D = A(16x32) * B(32x16) + C
    return __builtin_amdgcn_wmma_f32_16x16x32_bf16(false, a, false, b,
                                                   (short)0, c, false, false);
}

#define LDP 40   // padded LDS row stride (bf16 elements), 80B rows -> 16B aligned chunks

// ======================= Kernel 1: QKV projection ==========================
// X[4096,768] fp32 @ W[768,2304] fp32 + bias  ->  Q,K (bf16 [BH,S,64]), V^T (bf16 [BH,64,S])
__global__ __launch_bounds__(256)
void qkv_gemm_kernel(const float* __restrict__ X,
                     const float* __restrict__ W,
                     const float* __restrict__ bias,
                     __bf16* __restrict__ Qb,
                     __bf16* __restrict__ Kb,
                     __bf16* __restrict__ Vt) {
    const int K = 768, N = 2304;
    __shared__ __bf16 As[128 * LDP];   // [row][k]
    __shared__ __bf16 Bs[128 * LDP];   // transposed: [n][k]

    const int m0 = (blockIdx.x / 18) * 128;
    const int n0 = (blockIdx.x % 18) * 128;

    const int tid  = threadIdx.x;
    const int wv   = tid >> 5;
    const int lane = tid & 31;
    const int hf   = lane >> 4;
    const int l16  = lane & 15;
    const int waveRow = (wv & 3) * 32;   // 4 waves along M
    const int waveCol = (wv >> 2) * 64;  // 2 waves along N

    v8f acc[2][4];
#pragma unroll
    for (int mi = 0; mi < 2; ++mi)
#pragma unroll
        for (int ni = 0; ni < 4; ++ni)
#pragma unroll
            for (int i = 0; i < 8; ++i) acc[mi][ni][i] = 0.0f;

    for (int k0 = 0; k0 < K; k0 += 32) {
        // stage A tile: 128 rows x 32 k, fp32 -> bf16
        {
            int r  = tid >> 3;
            int kk = (tid & 7) * 4;
#pragma unroll
            for (int rep = 0; rep < 4; ++rep, r += 32) {
                const float4 v = *(const float4*)(X + (size_t)(m0 + r) * K + k0 + kk);
                __bf16* dst = &As[r * LDP + kk];
                dst[0] = (__bf16)v.x; dst[1] = (__bf16)v.y;
                dst[2] = (__bf16)v.z; dst[3] = (__bf16)v.w;
            }
        }
        // stage B tile transposed: Bs[n][k] = W[k0+k][n0+n], fp32 -> bf16
        {
            int n = tid & 127;
            int ks = (tid >> 7) * 16;
#pragma unroll
            for (int kk = 0; kk < 16; ++kk)
                Bs[n * LDP + ks + kk] = (__bf16)W[(size_t)(k0 + ks + kk) * N + n0 + n];
        }
        __syncthreads();

        v16bf af[2], bf[4];
#pragma unroll
        for (int mi = 0; mi < 2; ++mi) {
            const __bf16* base = &As[(waveRow + mi * 16 + l16) * LDP];
#pragma unroll
            for (int e = 0; e < 16; ++e) af[mi][e] = base[16 * hf + e];
        }
#pragma unroll
        for (int ni = 0; ni < 4; ++ni) {
            const __bf16* base = &Bs[(waveCol + ni * 16 + l16) * LDP + 16 * hf];
#pragma unroll
            for (int e = 0; e < 16; ++e) bf[ni][e] = base[e];
        }
#pragma unroll
        for (int mi = 0; mi < 2; ++mi)
#pragma unroll
            for (int ni = 0; ni < 4; ++ni)
                acc[mi][ni] = wmma_bf16(af[mi], bf[ni], acc[mi][ni]);
        __syncthreads();
    }

    // epilogue: add bias, scatter to Q (scaled for exp2 softmax), K, V^T
    const float qscale = 0.125f * 1.44269504088896340736f;  // 1/sqrt(64) * log2(e)
#pragma unroll
    for (int mi = 0; mi < 2; ++mi) {
#pragma unroll
        for (int ni = 0; ni < 4; ++ni) {
            const int gn    = n0 + waveCol + ni * 16 + l16;
            const int which = gn / 768;       // 0=q, 1=k, 2=v
            const int cc    = gn % 768;
            const int h     = cc >> 6;
            const int d     = cc & 63;
            const float bv  = bias[gn];
#pragma unroll
            for (int v = 0; v < 8; ++v) {
                const int gm = m0 + waveRow + mi * 16 + v + 8 * hf;
                const int b  = gm >> 11;
                const int s  = gm & 2047;
                const size_t bh = (size_t)(b * 12 + h);
                const float val = acc[mi][ni][v] + bv;
                if (which == 0)      Qb[(bh * 2048 + s) * 64 + d] = (__bf16)(val * qscale);
                else if (which == 1) Kb[(bh * 2048 + s) * 64 + d] = (__bf16)val;
                else                 Vt[(bh * 64 + d) * 2048 + s] = (__bf16)val;
            }
        }
    }
}

// ======================= Kernel 2: flash attention =========================
// One wave owns 32 query rows; online softmax over 32-key blocks (causal).
__global__ __launch_bounds__(256)
void attn_kernel(const __bf16* __restrict__ Qb,
                 const __bf16* __restrict__ Kb,
                 const __bf16* __restrict__ Vt,
                 __bf16* __restrict__ Ob) {      // [4096, 768] bf16
    const int S = 2048, D = 64, C = 768, H = 12;
    __shared__ __bf16 Pbuf[8][2][16 * LDP];      // per wave, per row-tile P (16x32)

    const int tid  = threadIdx.x;
    const int wv   = tid >> 5;
    const int lane = tid & 31;
    const int hf   = lane >> 4;
    const int l16  = lane & 15;

    const int bh  = blockIdx.x >> 3;
    const int b   = bh / H;
    const int h   = bh % H;
    const int qr0 = (blockIdx.x & 7) * 256 + wv * 32;

    const __bf16* Qp = Qb + (size_t)bh * S * D;
    const __bf16* Kp = Kb + (size_t)bh * S * D;
    const __bf16* Vp = Vt + (size_t)bh * D * S;

    // Q fragments (A layout), resident for whole kernel
    v16bf qf[2][2];
#pragma unroll
    for (int rt = 0; rt < 2; ++rt) {
        const __bf16* qrow = Qp + (size_t)(qr0 + rt * 16 + l16) * D;
#pragma unroll
        for (int kt = 0; kt < 2; ++kt) {
#pragma unroll
            for (int e = 0; e < 16; ++e) qf[rt][kt][e] = qrow[kt * 32 + 16 * hf + e];
        }
    }

    v8f o[2][4];
    float mrow[2][8], lrow[2][8];
#pragma unroll
    for (int rt = 0; rt < 2; ++rt) {
#pragma unroll
        for (int nt = 0; nt < 4; ++nt)
#pragma unroll
            for (int i = 0; i < 8; ++i) o[rt][nt][i] = 0.0f;
#pragma unroll
        for (int v = 0; v < 8; ++v) { mrow[rt][v] = -3.0e38f; lrow[rt][v] = 0.0f; }
    }

    const int nkb = (qr0 >> 5) + 1;    // causal: key blocks 0 .. qr0/32
    for (int kb = 0; kb < nkb; ++kb) {
        const int t0   = kb * 32;
        const bool dia = (kb == nkb - 1);

        // K^T fragments (B layout: sequential K per half-wave)
        v16bf kf[2][2];
#pragma unroll
        for (int ct = 0; ct < 2; ++ct) {
            const __bf16* krow = Kp + (size_t)(t0 + ct * 16 + l16) * D;
#pragma unroll
            for (int kt = 0; kt < 2; ++kt)
#pragma unroll
                for (int e = 0; e < 16; ++e) kf[ct][kt][e] = krow[kt * 32 + 16 * hf + e];
        }

        // scores = Q @ K^T   (already includes 1/sqrt(D)*log2e via Q scaling)
        v8f sc[2][2];
#pragma unroll
        for (int rt = 0; rt < 2; ++rt)
#pragma unroll
            for (int ct = 0; ct < 2; ++ct) {
#pragma unroll
                for (int i = 0; i < 8; ++i) sc[rt][ct][i] = 0.0f;
                sc[rt][ct] = wmma_bf16(qf[rt][0], kf[ct][0], sc[rt][ct]);
                sc[rt][ct] = wmma_bf16(qf[rt][1], kf[ct][1], sc[rt][ct]);
            }

        if (dia) {  // causal mask only on the diagonal block
#pragma unroll
            for (int rt = 0; rt < 2; ++rt)
#pragma unroll
                for (int ct = 0; ct < 2; ++ct)
#pragma unroll
                    for (int v = 0; v < 8; ++v) {
                        const int gq = qr0 + rt * 16 + v + 8 * hf;
                        const int gt = t0 + ct * 16 + l16;
                        if (gt > gq) sc[rt][ct][v] = -3.0e38f;
                    }
        }

        // online softmax update (row stats live in half-waves: xor 1,2,4,8)
#pragma unroll
        for (int rt = 0; rt < 2; ++rt) {
            float alpha[8];
#pragma unroll
            for (int v = 0; v < 8; ++v) {
                float mx = fmaxf(sc[rt][0][v], sc[rt][1][v]);
#pragma unroll
                for (int msk = 1; msk < 16; msk <<= 1)
                    mx = fmaxf(mx, __shfl_xor(mx, msk, 32));
                const float mnew = fmaxf(mrow[rt][v], mx);
                alpha[v] = exp2f(mrow[rt][v] - mnew);
                mrow[rt][v] = mnew;
                const float p0 = exp2f(sc[rt][0][v] - mnew);
                const float p1 = exp2f(sc[rt][1][v] - mnew);
                sc[rt][0][v] = p0; sc[rt][1][v] = p1;
                float rs = p0 + p1;
#pragma unroll
                for (int msk = 1; msk < 16; msk <<= 1)
                    rs += __shfl_xor(rs, msk, 32);
                lrow[rt][v] = lrow[rt][v] * alpha[v] + rs;
            }
            // P: C-layout -> LDS (row-major 16x32), to be re-read in A layout
            __bf16* pb = Pbuf[wv][rt];
#pragma unroll
            for (int ct = 0; ct < 2; ++ct)
#pragma unroll
                for (int v = 0; v < 8; ++v)
                    pb[(v + 8 * hf) * LDP + ct * 16 + l16] = (__bf16)sc[rt][ct][v];
            // rescale running output
#pragma unroll
            for (int nt = 0; nt < 4; ++nt)
#pragma unroll
                for (int v = 0; v < 8; ++v) o[rt][nt][v] *= alpha[v];
        }
        // in-wave LDS write->read ordering (no block barrier: waves diverge in kb)
        asm volatile("s_wait_dscnt 0" ::: "memory");

        // V^T fragments (B layout; contiguous thanks to [d][t] storage)
        v16bf vf[4];
#pragma unroll
        for (int nt = 0; nt < 4; ++nt) {
            const __bf16* vrow = Vp + (size_t)(nt * 16 + l16) * S + t0 + 16 * hf;
#pragma unroll
            for (int e = 0; e < 16; ++e) vf[nt][e] = vrow[e];
        }
        // O += P @ V
#pragma unroll
        for (int rt = 0; rt < 2; ++rt) {
            const __bf16* pr = &Pbuf[wv][rt][l16 * LDP];
            v16bf pf;
#pragma unroll
            for (int e = 0; e < 16; ++e) pf[e] = pr[16 * hf + e];
#pragma unroll
            for (int nt = 0; nt < 4; ++nt)
                o[rt][nt] = wmma_bf16(pf, vf[nt], o[rt][nt]);
        }
    }

    // normalize and write O in [B*S, C] layout for the proj GEMM
#pragma unroll
    for (int rt = 0; rt < 2; ++rt)
#pragma unroll
        for (int nt = 0; nt < 4; ++nt) {
            const int d = nt * 16 + l16;
#pragma unroll
            for (int v = 0; v < 8; ++v) {
                const int s = qr0 + rt * 16 + v + 8 * hf;
                Ob[((size_t)(b * S + s)) * C + h * 64 + d] =
                    (__bf16)(o[rt][nt][v] / lrow[rt][v]);
            }
        }
}

// ======================= Kernel 3: output projection =======================
// A[4096,768] bf16 @ W[768,768] fp32 + bias -> Out fp32
__global__ __launch_bounds__(256)
void proj_gemm_kernel(const __bf16* __restrict__ A,
                      const float* __restrict__ W,
                      const float* __restrict__ bias,
                      float* __restrict__ Out) {
    const int K = 768, N = 768;
    __shared__ __bf16 As[128 * LDP];
    __shared__ __bf16 Bs[128 * LDP];

    const int m0 = (blockIdx.x / 6) * 128;
    const int n0 = (blockIdx.x % 6) * 128;

    const int tid  = threadIdx.x;
    const int wv   = tid >> 5;
    const int lane = tid & 31;
    const int hf   = lane >> 4;
    const int l16  = lane & 15;
    const int waveRow = (wv & 3) * 32;
    const int waveCol = (wv >> 2) * 64;

    v8f acc[2][4];
#pragma unroll
    for (int mi = 0; mi < 2; ++mi)
#pragma unroll
        for (int ni = 0; ni < 4; ++ni)
#pragma unroll
            for (int i = 0; i < 8; ++i) acc[mi][ni][i] = 0.0f;

    for (int k0 = 0; k0 < K; k0 += 32) {
        {
            int r  = tid >> 3;
            int kk = (tid & 7) * 4;
#pragma unroll
            for (int rep = 0; rep < 4; ++rep, r += 32) {
                const __bf16* src = A + (size_t)(m0 + r) * K + k0 + kk;
                __bf16* dst = &As[r * LDP + kk];
#pragma unroll
                for (int j = 0; j < 4; ++j) dst[j] = src[j];
            }
        }
        {
            int n = tid & 127;
            int ks = (tid >> 7) * 16;
#pragma unroll
            for (int kk = 0; kk < 16; ++kk)
                Bs[n * LDP + ks + kk] = (__bf16)W[(size_t)(k0 + ks + kk) * N + n0 + n];
        }
        __syncthreads();

        v16bf af[2], bf[4];
#pragma unroll
        for (int mi = 0; mi < 2; ++mi) {
            const __bf16* base = &As[(waveRow + mi * 16 + l16) * LDP];
#pragma unroll
            for (int e = 0; e < 16; ++e) af[mi][e] = base[16 * hf + e];
        }
#pragma unroll
        for (int ni = 0; ni < 4; ++ni) {
            const __bf16* base = &Bs[(waveCol + ni * 16 + l16) * LDP + 16 * hf];
#pragma unroll
            for (int e = 0; e < 16; ++e) bf[ni][e] = base[e];
        }
#pragma unroll
        for (int mi = 0; mi < 2; ++mi)
#pragma unroll
            for (int ni = 0; ni < 4; ++ni)
                acc[mi][ni] = wmma_bf16(af[mi], bf[ni], acc[mi][ni]);
        __syncthreads();
    }

#pragma unroll
    for (int mi = 0; mi < 2; ++mi)
#pragma unroll
        for (int ni = 0; ni < 4; ++ni) {
            const int gn   = n0 + waveCol + ni * 16 + l16;
            const float bv = bias[gn];
#pragma unroll
            for (int v = 0; v < 8; ++v) {
                const int gm = m0 + waveRow + mi * 16 + v + 8 * hf;
                Out[(size_t)gm * N + gn] = acc[mi][ni][v] + bv;
            }
        }
}

// ============================== launcher ===================================
extern "C" void kernel_launch(void* const* d_in, const int* in_sizes, int n_in,
                              void* d_out, int out_size, void* d_ws, size_t ws_size,
                              hipStream_t stream) {
    const float* x      = (const float*)d_in[0];   // [2,2048,768]
    const float* W_attn = (const float*)d_in[1];   // [768,2304]
    const float* b_attn = (const float*)d_in[2];   // [2304]
    const float* W_proj = (const float*)d_in[3];   // [768,768]
    const float* b_proj = (const float*)d_in[4];   // [768]
    float* out = (float*)d_out;                    // [2,2048,768] fp32

    const size_t nQKV = (size_t)2 * 12 * 2048 * 64;  // 3,145,728 elems per buffer
    __bf16* Qb = (__bf16*)d_ws;          // pre-scaled q
    __bf16* Kb = Qb + nQKV;
    __bf16* Vt = Kb + nQKV;              // transposed v
    __bf16* Ob = Vt + nQKV;              // attention output, [4096,768]

    qkv_gemm_kernel<<<dim3(32 * 18), dim3(256), 0, stream>>>(x, W_attn, b_attn, Qb, Kb, Vt);
    attn_kernel<<<dim3(2 * 12 * 8), dim3(256), 0, stream>>>(Qb, Kb, Vt, Ob);
    proj_gemm_kernel<<<dim3(32 * 6), dim3(256), 0, stream>>>(Ob, W_proj, b_proj, out);
}
